// ProtoNet_16947940950251
// MI455X (gfx1250) — compile-verified, matrix operand order checked by reading, not run on hardware
//
#include <hip/hip_runtime.h>
#include <hip/hip_bf16.h>

// ---------------------------------------------------------------------------
// ProtoNet head for MI455X (gfx1250, wave32, WMMA).
//   logits = 2 * (Q @ P^T) - ||p||^2      (q^2 cancels in log_softmax)
//   out    = log_softmax(logits, axis=1)
// GEMM runs on v_wmma_f32_16x16x32_f16 with fp32->f16 (hi+lo) split for
// near-fp32 accuracy (3 WMMA accumulations per 32-wide k-tile).
// Wave tile 32x32 -> 4 independent accumulator chains so the WMMA pipe
// is not stalled on its own accumulator (round-1 disasm showed a single
// dependent chain with s_delay_alu TRANS32_DEP_1 stalls).
// ---------------------------------------------------------------------------

typedef __attribute__((ext_vector_type(16))) _Float16 v16h;
typedef __attribute__((ext_vector_type(8)))  _Float16 v8h;
typedef __attribute__((ext_vector_type(8)))  float    v8f;

#define NQ   8192   // queries
#define NS   2560   // supports
#define NC   512    // classes
#define DIM  1600   // 64*5*5

#define BM   128    // block tile M
#define BN   64     // block tile N
#define BK   32     // k tile (WMMA f16 K)
#define LDS_STRIDE 40   // BK + 8 halfs padding -> conflict-free row gather

union Frag16 { v16h v; v8h h[2]; };

// ---------------------------------------------------------------------------
// Kernel 1: class-mean prototypes + ||p||^2.  One block per class.
// Deterministic: thread 0 scans labels in order; fixed-order tree reduction.
// ---------------------------------------------------------------------------
__global__ void proto_prep_kernel(const float* __restrict__ support,
                                  const int*   __restrict__ label,
                                  float* __restrict__ protos,
                                  float* __restrict__ p2) {
    const int c = blockIdx.x;                 // class id
    __shared__ int   s_idx[32];
    __shared__ int   s_cnt;
    __shared__ float s_red[256];

    if (threadIdx.x == 0) {
        int n = 0;
        for (int i = 0; i < NS; ++i)
            if (label[i] == c && n < 32) s_idx[n++] = i;
        s_cnt = n;
    }
    __syncthreads();

    const int   n   = s_cnt;
    const float inv = 1.0f / (float)(n < 1 ? 1 : n);

    float p2_part = 0.0f;
    for (int d = threadIdx.x; d < DIM; d += blockDim.x) {
        float s = 0.0f;
        for (int j = 0; j < n; ++j)           // fixed order -> deterministic
            s += support[(size_t)s_idx[j] * DIM + d];
        const float pv = s * inv;
        protos[(size_t)c * DIM + d] = pv;
        p2_part += pv * pv;
    }

    s_red[threadIdx.x] = p2_part;
    __syncthreads();
    for (int off = 128; off > 0; off >>= 1) {
        if ((int)threadIdx.x < off) s_red[threadIdx.x] += s_red[threadIdx.x + off];
        __syncthreads();
    }
    if (threadIdx.x == 0) p2[c] = s_red[0];
}

// ---------------------------------------------------------------------------
// Kernel 2: logits = 2 * Q @ P^T - p2, via WMMA f16 hi/lo split emulation.
// Block = 256 threads = 8 waves (4 along M x 2 along N).
// Block tile 128x64; each wave owns a 32x32 tile (2x2 WMMA tiles).
// Per k-tile: 8 fragment b128 LDS loads feed 12 WMMAs across 4 acc chains.
// ---------------------------------------------------------------------------
__global__ void __launch_bounds__(256)
proto_gemm_kernel(const float* __restrict__ q,       // [NQ, DIM]
                  const float* __restrict__ protos,  // [NC, DIM]
                  const float* __restrict__ p2,      // [NC]
                  float* __restrict__ out) {         // [NQ, NC] logits
    __shared__ _Float16 Ah[BM][LDS_STRIDE];
    __shared__ _Float16 Al[BM][LDS_STRIDE];
    __shared__ _Float16 Bh[BN][LDS_STRIDE];
    __shared__ _Float16 Bl[BN][LDS_STRIDE];

    const int tid  = threadIdx.x;
    const int wid  = tid >> 5;        // wave 0..7
    const int lane = tid & 31;
    const int mT   = wid & 3;         // 4 wave tiles along M (32 rows each)
    const int nT   = wid >> 2;        // 2 wave tiles along N (32 cols each)
    const int m0   = blockIdx.y * BM; // block origin in M
    const int n0   = blockIdx.x * BN; // block origin in N

    // cooperative A load: 128 rows x 32 k, 16 floats/thread (4x float4)
    const int ar  = tid >> 1;         // 0..127
    const int ak0 = (tid & 1) * 16;   // 0 or 16
    const float* aptr = q + (size_t)(m0 + ar) * DIM + ak0;
    // cooperative B load: 64 rows x 32 k, 8 floats/thread (2x float4)
    const int br  = tid >> 2;         // 0..63
    const int bk0 = (tid & 3) * 8;    // 0,8,16,24
    const float* bptr = protos + (size_t)(n0 + br) * DIM + bk0;

    // WMMA fragment addressing (ISA 7.12.2 16-bit A/B layout):
    // lanes 0-15 hold K {0..7, 16..23}; lanes 16-31 hold K {8..15, 24..31}
    const int fr   = lane & 15;             // row/col within a 16x16 tile
    const int koff = (lane >> 4) * 8;       // 0 or 8

    v8f acc[2][2] = {};

#pragma unroll 1
    for (int kb = 0; kb < DIM; kb += BK) {
        __syncthreads();   // previous tile fully consumed

        // ---- global -> registers -> f16 hi/lo split -> LDS -------------
        float4 qa0 = *(const float4*)(aptr + kb);
        float4 qa1 = *(const float4*)(aptr + kb + 4);
        float4 qa2 = *(const float4*)(aptr + kb + 8);
        float4 qa3 = *(const float4*)(aptr + kb + 12);
        float4 pb0 = *(const float4*)(bptr + kb);
        float4 pb1 = *(const float4*)(bptr + kb + 4);

        if (kb + BK < DIM) {                  // prefetch next k-tile
            __builtin_prefetch(aptr + kb + BK, 0, 2);
            __builtin_prefetch(bptr + kb + BK, 0, 2);
        }

        float av[16] = { qa0.x, qa0.y, qa0.z, qa0.w, qa1.x, qa1.y, qa1.z, qa1.w,
                         qa2.x, qa2.y, qa2.z, qa2.w, qa3.x, qa3.y, qa3.z, qa3.w };
#pragma unroll
        for (int j = 0; j < 16; ++j) {
            _Float16 hi = (_Float16)av[j];
            _Float16 lo = (_Float16)(av[j] - (float)hi);
            Ah[ar][ak0 + j] = hi;
            Al[ar][ak0 + j] = lo;
        }
        float bv[8] = { pb0.x, pb0.y, pb0.z, pb0.w, pb1.x, pb1.y, pb1.z, pb1.w };
#pragma unroll
        for (int j = 0; j < 8; ++j) {
            _Float16 hi = (_Float16)bv[j];
            _Float16 lo = (_Float16)(bv[j] - (float)hi);
            Bh[br][bk0 + j] = hi;
            Bl[br][bk0 + j] = lo;
        }
        __syncthreads();

        // ---- build fragments (16B-aligned ds_load_b128 chunks) ---------
        Frag16 fah[2], fal[2], fbh[2], fbl[2];
#pragma unroll
        for (int i = 0; i < 2; ++i) {
            const int arow = mT * 32 + i * 16 + fr;
            fah[i].h[0] = *(const v8h*)&Ah[arow][koff];
            fah[i].h[1] = *(const v8h*)&Ah[arow][koff + 16];
            fal[i].h[0] = *(const v8h*)&Al[arow][koff];
            fal[i].h[1] = *(const v8h*)&Al[arow][koff + 16];
        }
#pragma unroll
        for (int j = 0; j < 2; ++j) {
            const int bcol = nT * 32 + j * 16 + fr;
            fbh[j].h[0] = *(const v8h*)&Bh[bcol][koff];
            fbh[j].h[1] = *(const v8h*)&Bh[bcol][koff + 16];
            fbl[j].h[0] = *(const v8h*)&Bl[bcol][koff];
            fbl[j].h[1] = *(const v8h*)&Bl[bcol][koff + 16];
        }

        // ---- 3-term split-precision accumulation, 4 independent chains -
#pragma unroll
        for (int i = 0; i < 2; ++i)
#pragma unroll
            for (int j = 0; j < 2; ++j) {
                acc[i][j] = __builtin_amdgcn_wmma_f32_16x16x32_f16(
                    false, fah[i].v, false, fbh[j].v, (short)0, acc[i][j], false, false);
                acc[i][j] = __builtin_amdgcn_wmma_f32_16x16x32_f16(
                    false, fah[i].v, false, fbl[j].v, (short)0, acc[i][j], false, false);
                acc[i][j] = __builtin_amdgcn_wmma_f32_16x16x32_f16(
                    false, fal[i].v, false, fbh[j].v, (short)0, acc[i][j], false, false);
            }
    }

    // ---- epilogue: logits = 2*dot - p2[col] ---------------------------
    const int mhi = (lane >> 4) * 8;          // C/D: VGPR e -> M = e (+8 hi lanes)
#pragma unroll
    for (int j = 0; j < 2; ++j) {
        const int   gcol  = n0 + nT * 32 + j * 16 + fr;   // N = lane&15
        const float pcol2 = p2[gcol];
#pragma unroll
        for (int i = 0; i < 2; ++i) {
            const int rbase = m0 + mT * 32 + i * 16 + mhi;
#pragma unroll
            for (int e = 0; e < 8; ++e) {
                out[(size_t)(rbase + e) * NC + gcol] = 2.0f * acc[i][j][e] - pcol2;
            }
        }
    }
}

// ---------------------------------------------------------------------------
// Kernel 3: in-place row-wise log_softmax over 512 columns. One block/row.
// ---------------------------------------------------------------------------
__global__ void logsoftmax_kernel(float* __restrict__ out) {
    const int r = blockIdx.x;
    float* row = out + (size_t)r * NC;
    __shared__ float s_red[256];
    const int t = threadIdx.x;

    const float a = row[t];
    const float b = row[t + 256];

    // max reduction (fixed order)
    s_red[t] = fmaxf(a, b);
    __syncthreads();
    for (int off = 128; off > 0; off >>= 1) {
        if (t < off) s_red[t] = fmaxf(s_red[t], s_red[t + off]);
        __syncthreads();
    }
    const float mx = s_red[0];
    __syncthreads();

    // sum of exp (fixed order)
    const float ea = __expf(a - mx);
    const float eb = __expf(b - mx);
    s_red[t] = ea + eb;
    __syncthreads();
    for (int off = 128; off > 0; off >>= 1) {
        if (t < off) s_red[t] += s_red[t + off];
        __syncthreads();
    }
    const float lse = __logf(s_red[0]);
    __syncthreads();

    row[t]       = a - mx - lse;
    row[t + 256] = b - mx - lse;
}

// ---------------------------------------------------------------------------
extern "C" void kernel_launch(void* const* d_in, const int* in_sizes, int n_in,
                              void* d_out, int out_size, void* d_ws, size_t ws_size,
                              hipStream_t stream) {
    const float* support = (const float*)d_in[0];   // [2560, 64, 5, 5]
    const float* query   = (const float*)d_in[1];   // [8192, 64, 5, 5]
    const int*   label   = (const int*)d_in[2];     // [2560]
    float*       out     = (float*)d_out;           // [8192, 512]

    float* protos = (float*)d_ws;                   // [512, 1600]
    float* p2     = protos + (size_t)NC * DIM;      // [512]

    // 1) prototypes + ||p||^2
    proto_prep_kernel<<<NC, 256, 0, stream>>>(support, label, protos, p2);

    // 2) WMMA GEMM -> logits
    dim3 grid(NC / BN, NQ / BM);                    // (8, 64)
    proto_gemm_kernel<<<grid, 256, 0, stream>>>(query, protos, p2, out);

    // 3) row-wise log_softmax in place
    logsoftmax_kernel<<<NQ, 256, 0, stream>>>(out);
}
